// CRFLoss_32899449487478
// MI455X (gfx1250) — matrix operand order repaired and checked by reading, not compiled, not run on hardware
//
#include <hip/hip_runtime.h>

typedef __attribute__((ext_vector_type(16))) _Float16 v16h;
typedef __attribute__((ext_vector_type(8)))  _Float16 v8h;
typedef __attribute__((ext_vector_type(8)))  float    v8f;

#define CRF_N 2048
#define CRF_T 200
#define CRF_L 129
#define LPK   160   // padded prev-tag (K) dim: 5 chunks of 32
#define LPJ   144   // padded next-tag (J) dim: 9 tiles of 16
#define NKC   5
#define NJT   9
#define NEG_BIG (-1.0e30f)
#define Q_EPS  (1.0e-37f)
#define LOG2E 1.4426950408889634f
#define LN2   0.6931471805599453f

// Fast exp2/log2 lowering to single v_exp_f32 / v_log_f32 where available.
#if __has_builtin(__builtin_amdgcn_exp2f)
__device__ __forceinline__ float fast_exp2(float x) { return __builtin_amdgcn_exp2f(x); }
#else
__device__ __forceinline__ float fast_exp2(float x) { return exp2f(x); }
#endif
#if __has_builtin(__builtin_amdgcn_logf)
__device__ __forceinline__ float fast_log2(float x) { return __builtin_amdgcn_logf(x); }
#else
__device__ __forceinline__ float fast_log2(float x) { return log2f(x); }
#endif

// ---------------------------------------------------------------------------
// Kernel 1: numerator per sequence -> ws_num[n]
// One wave per sequence; lanes stride over time steps.
// ---------------------------------------------------------------------------
__global__ __launch_bounds__(256) void crf_num_kernel(
    const float* __restrict__ score, const float* __restrict__ trans,
    const float* __restrict__ startT, const float* __restrict__ endT,
    const int* __restrict__ gold, const int* __restrict__ lengths,
    float* __restrict__ ws_num)
{
  const int wavesPerBlock = blockDim.x >> 5;
  const int n    = blockIdx.x * wavesPerBlock + (threadIdx.x >> 5);
  const int lane = threadIdx.x & 31;
  if (n >= CRF_N) return;

  const int len = lengths[n];
  float s = 0.0f;
  for (int t = lane; t < CRF_T; t += 32) {
    if (t < len) {
      const int g = gold[n * CRF_T + t];
      s += score[((long)n * CRF_T + t) * CRF_L + g];
      if (t >= 1) {
        const int gp = gold[n * CRF_T + t - 1];
        s += trans[gp * CRF_L + g];
      }
    }
  }
  #pragma unroll
  for (int m = 16; m >= 1; m >>= 1) s += __shfl_xor(s, m, 32);

  if (lane == 0) {
    const int g0 = gold[n * CRF_T];
    const int gl = gold[n * CRF_T + (len - 1)];
    ws_num[n] = s + startT[g0] + endT[gl];
  }
}

// ---------------------------------------------------------------------------
// Kernel 2: forward algorithm (denominator) via WMMA in exp space.
// One wave per workgroup; each wave owns 16 sequences.
// alpha lives in WMMA C/D f32 layout (9 x v8f). Per step:
//   m = rowmax(alpha); P = f16(exp2((alpha-m)*log2e)) -> LDS;
//   Q = P x E        (v_wmma_f32_16x16x32_f16, E = exp(Trans) in LDS);
//   alpha += maskf * ((m + log2(max(Q,eps))*ln2 + logit) - alpha)   [branch-free]
// All values kept finite (NEG_BIG sentinels + Q_EPS clamp) so the arithmetic
// mask blend is NaN-free and the compiler cannot re-branch the log.
// ---------------------------------------------------------------------------
__global__ __launch_bounds__(32) void crf_forward_kernel(
    const float* __restrict__ score, const float* __restrict__ trans,
    const float* __restrict__ startT, const float* __restrict__ endT,
    const int* __restrict__ lengths, const float* __restrict__ ws_num,
    float* __restrict__ ws_part)
{
  __shared__ __align__(32) _Float16 sET[LPJ * LPK];   // ET[j][i] = exp(Trans[i][j])
  __shared__ __align__(32) _Float16 sP[16 * LPK];     // P = exp(alpha - m), row-major [M][i]

  const int lane = threadIdx.x & 31;
  const int half = lane >> 4;   // 0: rows M=0..7 ; 1: rows M=8..15
  const int l15  = lane & 15;

  // --- build E^T (f16, zero-padded) once ------------------------------------
  for (int idx = lane; idx < LPJ * LPK; idx += 32) {
    const int j = idx / LPK, i = idx % LPK;
    float v = 0.0f;
    if (i < CRF_L && j < CRF_L) v = fast_exp2(trans[i * CRF_L + j] * LOG2E);
    sET[j * LPK + i] = (_Float16)v;
  }
  // zero the never-rewritten pad columns of P (i = 144..159)
  for (int idx = lane; idx < 16 * (LPK - LPJ); idx += 32) {
    const int row = idx / (LPK - LPJ), c = LPJ + idx % (LPK - LPJ);
    sP[row * LPK + c] = (_Float16)0.0f;
  }
  __syncthreads();

  const int n0 = blockIdx.x * 16;

  // per-lane: lengths and score row base pointers for the 8 rows this half owns
  int len8[8];
  const float* base8[8];
  #pragma unroll
  for (int r = 0; r < 8; r++) {
    const int n = n0 + r + 8 * half;
    len8[r]  = lengths[n];
    base8[r] = score + (size_t)n * CRF_T * CRF_L;
  }

  // --- alpha0 = start + score[:,0,:] (D-layout registers) -------------------
  v8f alpha[NJT];
  #pragma unroll
  for (int jt = 0; jt < NJT; jt++) {
    const int col  = jt * 16 + l15;
    const int colc = (col < CRF_L) ? col : (CRF_L - 1);
    const float st = startT[colc];
    #pragma unroll
    for (int r = 0; r < 8; r++) {
      const float v = st + base8[r][colc];
      alpha[jt][r] = (col < CRF_L) ? v : NEG_BIG;
    }
  }

  // --- recursion over time --------------------------------------------------
  for (int t = 1; t < CRF_T; t++) {
    const size_t toff = (size_t)t * CRF_L;

    // 0) per-row mask as arithmetic blend factor (computed once per step)
    float maskf8[8];
    #pragma unroll
    for (int r = 0; r < 8; r++) maskf8[r] = (t < len8[r]) ? 1.0f : 0.0f;

    // 1) per-row max over all 144 tag columns (within-half shuffle reduce)
    float m[8];
    #pragma unroll
    for (int r = 0; r < 8; r++) {
      float mx = alpha[0][r];
      #pragma unroll
      for (int jt = 1; jt < NJT; jt++) mx = fmaxf(mx, alpha[jt][r]);
      mx = fmaxf(mx, __shfl_xor(mx, 1, 32));
      mx = fmaxf(mx, __shfl_xor(mx, 2, 32));
      mx = fmaxf(mx, __shfl_xor(mx, 4, 32));
      mx = fmaxf(mx, __shfl_xor(mx, 8, 32));
      m[r] = mx;
    }

    // 2) P = f16(exp(alpha - m)) -> LDS, row-major [M][i]
    #pragma unroll
    for (int jt = 0; jt < NJT; jt++) {
      const int col = jt * 16 + l15;
      #pragma unroll
      for (int r = 0; r < 8; r++) {
        const int M = r + 8 * half;
        sP[M * LPK + col] = (_Float16)fast_exp2((alpha[jt][r] - m[r]) * LOG2E);
      }
    }
    // same-wave LDS is in-order: no barrier needed before reading back

    // 3) A fragments (16-bit A layout: lanes 0-15 K{0..7,16..23}, 16-31 K{8..15,24..31})
    v16h A[NKC];
    #pragma unroll
    for (int kc = 0; kc < NKC; kc++) {
      const int kb = kc * 32;
      const v8h a0 = *(const v8h*)&sP[l15 * LPK + kb + 8 * half];
      const v8h a1 = *(const v8h*)&sP[l15 * LPK + kb + 16 + 8 * half];
      #pragma unroll
      for (int e = 0; e < 8; e++) { A[kc][e] = a0[e]; A[kc][e + 8] = a1[e]; }
    }

    // 4) Q = P x E per 16-column tile; alpha blended branch-free
    #pragma unroll
    for (int jt = 0; jt < NJT; jt++) {
      v8f acc = {0.f, 0.f, 0.f, 0.f, 0.f, 0.f, 0.f, 0.f};
      #pragma unroll
      for (int kc = 0; kc < NKC; kc++) {
        const int kb = kc * 32;
        const v8h b0 = *(const v8h*)&sET[(jt * 16 + l15) * LPK + kb + 8 * half];
        const v8h b1 = *(const v8h*)&sET[(jt * 16 + l15) * LPK + kb + 16 + 8 * half];
        v16h B;
        #pragma unroll
        for (int e = 0; e < 8; e++) { B[e] = b0[e]; B[e + 8] = b1[e]; }
        acc = __builtin_amdgcn_wmma_f32_16x16x32_f16(
            /*neg_a=*/false, A[kc], /*neg_b=*/false, B,
            /*c_mod=*/(short)0, acc, /*reuse_a=*/false, /*reuse_b=*/false);
      }
      const int col  = jt * 16 + l15;
      const int colc = (col < CRF_L) ? col : (CRF_L - 1);   // clamp: load always valid
      #pragma unroll
      for (int r = 0; r < 8; r++) {
        float logit = base8[r][toff + colc];                 // unconditional coalesced load
        logit = (col < CRF_L) ? logit : NEG_BIG;             // cndmask, folds away for jt<8
        const float q  = fmaxf(acc[r], Q_EPS);               // keep log finite (pads: q=0)
        const float na = m[r] + fast_log2(q) * LN2 + logit;  // finite everywhere
        const float ao = alpha[jt][r];
        alpha[jt][r] = ao + maskf8[r] * (na - ao);           // NaN-free arithmetic blend
      }
      if (t + 1 < CRF_T)                                     // uniform scalar branch
        __builtin_prefetch(&base8[0][toff + CRF_L + colc], 0, 1);
    }
  }

  // --- log_denominator = logsumexp(alpha + end) per row ---------------------
  #pragma unroll
  for (int r = 0; r < 8; r++) {
    float v[NJT];
    float mx = NEG_BIG;
    #pragma unroll
    for (int jt = 0; jt < NJT; jt++) {
      const int col  = jt * 16 + l15;
      const int colc = (col < CRF_L) ? col : (CRF_L - 1);
      const float vv = alpha[jt][r] + endT[colc];
      v[jt] = (col < CRF_L) ? vv : NEG_BIG;
      mx = fmaxf(mx, v[jt]);
    }
    mx = fmaxf(mx, __shfl_xor(mx, 1, 32));
    mx = fmaxf(mx, __shfl_xor(mx, 2, 32));
    mx = fmaxf(mx, __shfl_xor(mx, 4, 32));
    mx = fmaxf(mx, __shfl_xor(mx, 8, 32));
    float s = 0.0f;
    #pragma unroll
    for (int jt = 0; jt < NJT; jt++) s += fast_exp2((v[jt] - mx) * LOG2E);
    s += __shfl_xor(s, 1, 32);
    s += __shfl_xor(s, 2, 32);
    s += __shfl_xor(s, 4, 32);
    s += __shfl_xor(s, 8, 32);
    if (l15 == 0) {
      const int n = n0 + r + 8 * half;
      ws_part[n] = ws_num[n] - (mx + fast_log2(s) * LN2);
    }
  }
}

// ---------------------------------------------------------------------------
// Kernel 3: deterministic fixed-order reduction of 2048 partials -> out[0]
// ---------------------------------------------------------------------------
__global__ __launch_bounds__(256) void crf_reduce_kernel(
    const float* __restrict__ part, float* __restrict__ out)
{
  __shared__ float buf[256];
  float s = 0.0f;
  for (int i = threadIdx.x; i < CRF_N; i += 256) s += part[i];
  buf[threadIdx.x] = s;
  __syncthreads();
  for (int k = 128; k > 0; k >>= 1) {
    if (threadIdx.x < k) buf[threadIdx.x] += buf[threadIdx.x + k];
    __syncthreads();
  }
  if (threadIdx.x == 0) out[0] = buf[0];
}

// ---------------------------------------------------------------------------
extern "C" void kernel_launch(void* const* d_in, const int* in_sizes, int n_in,
                              void* d_out, int out_size, void* d_ws, size_t ws_size,
                              hipStream_t stream) {
  const float* score  = (const float*)d_in[0];   // (N,T,L)
  const float* trans  = (const float*)d_in[1];   // (L,L)
  const float* startT = (const float*)d_in[2];   // (L,)
  const float* endT   = (const float*)d_in[3];   // (L,)
  const int*   gold   = (const int*)d_in[4];     // (N,T)
  const int*   lens   = (const int*)d_in[5];     // (N,)

  float* ws_num  = (float*)d_ws;          // N floats
  float* ws_part = ws_num + CRF_N;        // N floats
  float* out     = (float*)d_out;         // 1 float

  // numerator: one wave per sequence, 8 waves per block
  crf_num_kernel<<<CRF_N / 8, 256, 0, stream>>>(score, trans, startT, endT,
                                                gold, lens, ws_num);
  // forward recursion: one wave per workgroup, 16 sequences per wave, 128 WGs
  crf_forward_kernel<<<CRF_N / 16, 32, 0, stream>>>(
      score, trans, startT, endT, lens, ws_num, ws_part);
  // deterministic reduction
  crf_reduce_kernel<<<1, 256, 0, stream>>>(ws_part, out);
}
